// Attention_75093208203986
// MI455X (gfx1250) — matrix operand order repaired and checked by reading, not compiled, not run on hardware
//
#include <hip/hip_runtime.h>
#include <hip/hip_bf16.h>

// ---------------------------------------------------------------------------
// Types for CDNA5 WMMA (wave32): D(16x16,f32) = A(16x32,f16) x B(32x16,f16) + C
// ---------------------------------------------------------------------------
typedef _Float16 half_t;
typedef _Float16 h8_t  __attribute__((ext_vector_type(8)));   // 16 bytes
typedef _Float16 h16_t __attribute__((ext_vector_type(16)));  // 32 bytes (8 VGPRs)
typedef float    f8_t  __attribute__((ext_vector_type(8)));   // 8 VGPRs accumulator

union FragU { h16_t v; h8_t h8[2]; };

// A-matrix 16x32 f16 fragment from a row-major (token x dim) buffer.
// ISA layout: row = lane&15; lanes 0-15 hold K = k0+[0..7] (vgpr0-3) and
// K = k0+16+[0..7] (vgpr4-7); lanes 16-31 the same shifted by +8.
__device__ inline h16_t load_frag_a(const half_t* base, int ld, int row0, int k0,
                                    int lane) {
    int row = row0 + (lane & 15);
    int kk  = k0 + ((lane >> 4) << 3);           // + 8*laneHigh
    const half_t* p = base + row * ld + kk;
    FragU f;
    f.h8[0] = *(const h8_t*)(p);                 // K = kk .. kk+7
    f.h8[1] = *(const h8_t*)(p + 16);            // K = kk+16 .. kk+23
    return f.v;
}

// B-matrix 32x16 f16 fragment from an n-major (outCol x k) buffer (i.e. W^T).
// ISA layout: n = lane&15; lanes 0-15 hold K = k0+[0..7] and k0+16+[0..7]; lanes 16-31 the same shifted by +8.
__device__ inline h16_t load_frag_b(const half_t* base, int ld, int n0, int k0,
                                    int lane) {
    int n  = n0 + (lane & 15);
    int kk = k0 + ((lane >> 4) << 3);            // + 8*laneHigh
    const half_t* p = base + n * ld + kk;
    FragU f;
    f.h8[0] = *(const h8_t*)(p);
    f.h8[1] = *(const h8_t*)(p + 16);
    return f.v;
}

__device__ inline f8_t wmma16x16x32(h16_t a, h16_t b, f8_t c) {
    return __builtin_amdgcn_wmma_f32_16x16x32_f16(false, a, false, b,
                                                  (short)0, c, false, false);
}

#define ZERO8 {0.f, 0.f, 0.f, 0.f, 0.f, 0.f, 0.f, 0.f}

// ---------------------------------------------------------------------------
// Problem constants
// ---------------------------------------------------------------------------
#define BWIN 2048
#define NTOK 64
#define CDIM 256
#define NHEAD 8
#define DHEAD 32
#define NWIN 256
#define SCALE 0.17677669529663687f   // 32^-0.5

// LDS layout (halfs). Rows padded so every WMMA fragment chunk is 16B aligned.
#define LDQ 264                      // 64 x 264 f16
#define LDV 72                       // 256 x 72 f16 (v transposed: [h*32+d][token])
#define LDP 72                       // per-wave p stripe: 16 x 72 f16
#define SM_QH  0
#define SM_KH  (SM_QH + 64 * LDQ)
#define SM_VT  (SM_KH + 64 * LDQ)
#define SM_P   (SM_VT + 256 * LDV)
#define SM_XO  (SM_P  + 8 * 16 * LDP)          // x staging, reused as attn-out
#define SM_HALFS (SM_XO + 64 * LDQ)
#define SM_BYTES (SM_HALFS * 2)                 // 156,672 B < 320 KB WGP LDS

// Workspace layout (bytes): f16 transposed weights
#define WS_WQT 0
#define WS_WKVT (WS_WQT + 256 * 256 * 2)
#define WS_WPT  (WS_WKVT + 512 * 256 * 2)

// ---------------------------------------------------------------------------
// Kernel 0: fp32 weights -> f16, transposed to n-major; scale folded into Wq.
// ---------------------------------------------------------------------------
__global__ void prep_weights(const float* __restrict__ Wq,
                             const float* __restrict__ Wkv,
                             const float* __restrict__ Wp,
                             half_t* __restrict__ wqT,
                             half_t* __restrict__ wkvT,
                             half_t* __restrict__ wpT) {
    int i = blockIdx.x * blockDim.x + threadIdx.x;   // 262144 total
    if (i < 65536) {                                  // WqT [n=256][k=256]
        int n = i >> 8, k = i & 255;
        wqT[i] = (half_t)(Wq[k * 256 + n] * SCALE);
    } else if (i < 65536 + 131072) {                  // WkvT [n=512][k=256]
        int j = i - 65536;
        int n = j >> 8, k = j & 255;
        wkvT[j] = (half_t)(Wkv[k * 512 + n]);
    } else if (i < 262144) {                          // WpT [n=256][k=256]
        int j = i - 196608;
        int n = j >> 8, k = j & 255;
        wpT[j] = (half_t)(Wp[k * 256 + n]);
    }
}

// ---------------------------------------------------------------------------
// Kernel 1: fully fused per-window attention. 1 block = 1 window, 8 waves.
// ---------------------------------------------------------------------------
__global__ __launch_bounds__(256)
void swin_attn_fused(const float* __restrict__ x,
                     const float* __restrict__ mask,
                     const float* __restrict__ bq,
                     const float* __restrict__ bkv,
                     const float* __restrict__ bp,
                     const half_t* __restrict__ wqT,
                     const half_t* __restrict__ wkvT,
                     const half_t* __restrict__ wpT,
                     float* __restrict__ out) {
    extern __shared__ char smem_raw[];
    half_t* sm  = (half_t*)smem_raw;
    half_t* qh  = sm + SM_QH;    // [64][LDQ] scaled q, cols = h*32+d
    half_t* kh  = sm + SM_KH;    // [64][LDQ] k
    half_t* vth = sm + SM_VT;    // [256][LDV] v transposed: [h*32+d][token]
    half_t* pws = sm + SM_P;     // [8 waves][16][LDP] softmax probs stripe
    half_t* xh  = sm + SM_XO;    // [64][LDQ] x (f16); reused as attn output
    half_t* ohh = xh;

    const int b    = blockIdx.x;
    const int tid  = threadIdx.x;
    const int wave = tid >> 5;
    const int lane = tid & 31;
    const int lh8  = (lane >> 4) << 3;   // 8 * laneHigh (C/D row offset)
    const int ncl  = lane & 15;          // C/D column within tile

    // ---- Step 1: stage x[b] (64x256 fp32) into LDS as f16 (16B LDS stores) --
    const float* xb = x + (size_t)b * NTOK * CDIM;
    for (int i = tid; i < NTOK * CDIM / 8; i += 256) {
        const float4* src = (const float4*)xb + i * 2;
        float4 f0 = src[0];
        float4 f1 = src[1];
        int row = (i * 8) >> 8;
        int col = (i * 8) & 255;
        h8_t p;
        p[0] = (half_t)f0.x; p[1] = (half_t)f0.y;
        p[2] = (half_t)f0.z; p[3] = (half_t)f0.w;
        p[4] = (half_t)f1.x; p[5] = (half_t)f1.y;
        p[6] = (half_t)f1.z; p[7] = (half_t)f1.w;
        *(h8_t*)(xh + row * LDQ + col) = p;
    }
    __syncthreads();

    // ---- Step 2: QKV projection. 64x768 out = 4 mt x 48 nt tiles of 16x16.
    // Wave w owns n-tiles [6w, 6w+6) x all 4 m-tiles: B-frag loaded once per
    // K-step feeds 4 WMMAs. gcol is wave-uniform -> scalar branches only.
    for (int ntl = 0; ntl < 6; ++ntl) {
        const int nt   = wave * 6 + ntl;
        const int gcol = nt << 4;                       // 0..752, uniform
        const half_t* wT;
        int wrow;
        if (gcol < 256) { wT = wqT;  wrow = gcol; }
        else            { wT = wkvT; wrow = gcol - 256; }

        f8_t acc[4] = {ZERO8, ZERO8, ZERO8, ZERO8};
#pragma unroll
        for (int k0 = 0; k0 < CDIM; k0 += 32) {
            h16_t bf = load_frag_b(wT, 256, wrow, k0, lane);
#pragma unroll
            for (int mt = 0; mt < 4; ++mt) {
                h16_t a = load_frag_a(xh, LDQ, mt * 16, k0, lane);
                acc[mt] = wmma16x16x32(a, bf, acc[mt]);
            }
        }

        if (gcol < 256) {                                // ---- q (pre-scaled)
            float bias = bq[gcol + ncl] * SCALE;
#pragma unroll
            for (int mt = 0; mt < 4; ++mt) {
                int mrow0 = mt * 16 + lh8;
#pragma unroll
                for (int r = 0; r < 8; ++r)
                    qh[(mrow0 + r) * LDQ + gcol + ncl] = (half_t)(acc[mt][r] + bias);
            }
        } else if (gcol < 512) {                         // ---- k
            int kc = gcol - 256;
            float bias = bkv[kc + ncl];
#pragma unroll
            for (int mt = 0; mt < 4; ++mt) {
                int mrow0 = mt * 16 + lh8;
#pragma unroll
                for (int r = 0; r < 8; ++r)
                    kh[(mrow0 + r) * LDQ + kc + ncl] = (half_t)(acc[mt][r] + bias);
            }
        } else {                                         // ---- v -> transposed
            int vc = gcol - 512 + ncl;                   // = h*32 + d
            float bias = bkv[gcol - 256 + ncl];
#pragma unroll
            for (int mt = 0; mt < 4; ++mt) {
                int mrow0 = mt * 16 + lh8;
                h8_t pack;
#pragma unroll
                for (int r = 0; r < 8; ++r) pack[r] = (half_t)(acc[mt][r] + bias);
                *(h8_t*)(vth + vc * LDV + mrow0) = pack; // 8 tokens contiguous
            }
        }
    }
    __syncthreads();

    // ---- Step 3: attention. wave == head. dh=32 => single-K WMMA for QK^T --
    {
        const int h = wave;
        half_t* pst = pws + h * (16 * LDP);
        const float* mbase = mask + (size_t)(b & (NWIN - 1)) * NTOK * NTOK;

        for (int mt = 0; mt < 4; ++mt) {
            h16_t aq = load_frag_a(qh, LDQ, mt * 16, h * DHEAD, lane);
            f8_t s[4];
#pragma unroll
            for (int nt = 0; nt < 4; ++nt) {
                h16_t bk = load_frag_b(kh, LDQ, nt * 16, h * DHEAD, lane);
                f8_t z = ZERO8;
                s[nt] = wmma16x16x32(aq, bk, z);
            }
            // mask add + row softmax (row of 64 lives in 16 lanes x 4 tiles)
#pragma unroll
            for (int r = 0; r < 8; ++r) {
                int qrow = mt * 16 + lh8 + r;
                float sv[4];
                float rmax = -3.4e38f;
#pragma unroll
                for (int nt = 0; nt < 4; ++nt) {
                    sv[nt] = s[nt][r] + mbase[qrow * NTOK + nt * 16 + ncl];
                    rmax = fmaxf(rmax, sv[nt]);
                }
                for (int m = 1; m < 16; m <<= 1)
                    rmax = fmaxf(rmax, __shfl_xor(rmax, m, 32));
                float sum = 0.f;
#pragma unroll
                for (int nt = 0; nt < 4; ++nt) {
                    sv[nt] = __expf(sv[nt] - rmax);
                    sum += sv[nt];
                }
                for (int m = 1; m < 16; m <<= 1)
                    sum += __shfl_xor(sum, m, 32);
                float inv = 1.0f / sum;
                int prow = lh8 + r;
#pragma unroll
                for (int nt = 0; nt < 4; ++nt)
                    pst[prow * LDP + nt * 16 + ncl] = (half_t)(sv[nt] * inv);
            }
            // P (16x64) @ V (64x32): two K-steps, two N-tiles
            f8_t o0 = ZERO8;
            f8_t o1 = ZERO8;
#pragma unroll
            for (int kb = 0; kb < NTOK; kb += 32) {
                h16_t ap  = load_frag_a(pst, LDP, 0, kb, lane);
                h16_t bv0 = load_frag_b(vth, LDV, h * DHEAD, kb, lane);
                h16_t bv1 = load_frag_b(vth, LDV, h * DHEAD + 16, kb, lane);
                o0 = wmma16x16x32(ap, bv0, o0);
                o1 = wmma16x16x32(ap, bv1, o1);
            }
            int orow0 = mt * 16 + lh8;
#pragma unroll
            for (int r = 0; r < 8; ++r) {
                ohh[(orow0 + r) * LDQ + h * DHEAD + ncl]      = (half_t)o0[r];
                ohh[(orow0 + r) * LDQ + h * DHEAD + 16 + ncl] = (half_t)o1[r];
            }
        }
    }
    __syncthreads();

    // ---- Step 4: output projection  out = ohh @ Wp + bp  (fp32 store).
    // Wave w owns m-tile (w&3) x n-tiles [(w>>2)*8, +8): A-frag loaded once
    // per K-step feeds 8 WMMAs into 8 accumulators.
    {
        float* ob = out + (size_t)b * NTOK * CDIM;
        const int mt  = wave & 3;
        const int nt0 = (wave >> 2) << 3;                // 0 or 8

        f8_t acc[8] = {ZERO8, ZERO8, ZERO8, ZERO8, ZERO8, ZERO8, ZERO8, ZERO8};
#pragma unroll
        for (int k0 = 0; k0 < CDIM; k0 += 32) {
            h16_t a = load_frag_a(ohh, LDQ, mt * 16, k0, lane);
#pragma unroll
            for (int j = 0; j < 8; ++j) {
                h16_t bf = load_frag_b(wpT, 256, (nt0 + j) * 16, k0, lane);
                acc[j] = wmma16x16x32(a, bf, acc[j]);
            }
        }
        const int m0 = mt * 16 + lh8;
#pragma unroll
        for (int j = 0; j < 8; ++j) {
            int gn = (nt0 + j) * 16 + ncl;
            float bias = bp[gn];
#pragma unroll
            for (int r = 0; r < 8; ++r)
                ob[(m0 + r) * CDIM + gn] = acc[j][r] + bias;
        }
    }
}

// ---------------------------------------------------------------------------
// Host entry
// ---------------------------------------------------------------------------
extern "C" void kernel_launch(void* const* d_in, const int* in_sizes, int n_in,
                              void* d_out, int out_size, void* d_ws, size_t ws_size,
                              hipStream_t stream) {
    const float* x    = (const float*)d_in[0];
    const float* mask = (const float*)d_in[1];
    const float* Wq   = (const float*)d_in[2];
    const float* bq   = (const float*)d_in[3];
    const float* Wkv  = (const float*)d_in[4];
    const float* bkv  = (const float*)d_in[5];
    const float* Wp   = (const float*)d_in[6];
    const float* bp   = (const float*)d_in[7];
    float* out = (float*)d_out;

    char* ws = (char*)d_ws;
    half_t* wqT  = (half_t*)(ws + WS_WQT);
    half_t* wkvT = (half_t*)(ws + WS_WKVT);
    half_t* wpT  = (half_t*)(ws + WS_WPT);

    prep_weights<<<1024, 256, 0, stream>>>(Wq, Wkv, Wp, wqT, wkvT, wpT);

    swin_attn_fused<<<BWIN, 256, SM_BYTES, stream>>>(
        x, mask, bq, bkv, bp, wqT, wkvT, wpT, out);
}
